// Fp8AmpereLinear_6725918786085
// MI455X (gfx1250) — matrix operand order repaired and checked by reading, compiled, not run on hardware
//
#include <hip/hip_runtime.h>
#include <stdint.h>

// Problem constants (from reference): x[M,K] bf16, W[N,K] fp8-e4m3 bits in int32,
// weight_scale[N] bf16, bias[N] bf16 -> out[M,N] bf16.
#define M_DIM 4096
#define K_DIM 4096
#define N_DIM 12288

#define BM 256
#define BN 128
#define BK 32
#define LDS_STRIDE 40   // BK + 8 halfwords pad: keeps 16B alignment, spreads LDS banks
#define NTHREADS 512    // 16 waves (wave32)

#ifndef __has_builtin
#define __has_builtin(x) 0
#endif
#if __has_builtin(__builtin_amdgcn_global_load_async_to_lds_b128)
#define HAVE_ASYNC_LDS 1
#else
#define HAVE_ASYNC_LDS 0
#endif

typedef __attribute__((ext_vector_type(16))) __bf16 v16bf;
typedef __attribute__((ext_vector_type(8)))  float  v8f;

#if HAVE_ASYNC_LDS
// Builtin signature (from compiler diagnostic): params are pointers to
// 16-byte int vectors in AS1 (global) and AS3 (LDS), then imm offset, imm cpol.
typedef int v4i_ __attribute__((__vector_size__(16)));
typedef __attribute__((address_space(1))) v4i_ as1_v4i;
typedef __attribute__((address_space(3))) v4i_ as3_v4i;
#endif

__device__ __forceinline__ void wait_async0() {
#if __has_builtin(__builtin_amdgcn_s_wait_asynccnt)
  __builtin_amdgcn_s_wait_asynccnt(0);
#else
  asm volatile("s_wait_asynccnt 0x0" ::: "memory");
#endif
}

__device__ __forceinline__ float bf16_bits_to_f32(uint32_t u) {
  union { uint32_t i; float f; } c; c.i = u << 16; return c.f;
}
__device__ __forceinline__ uint16_t f32_to_bf16_bits(float f) {
  union { float f; uint32_t i; } c; c.f = f;
  uint32_t u = c.i;
  u += 0x7FFFu + ((u >> 16) & 1u);        // round to nearest even
  return (uint16_t)(u >> 16);
}
// FP8 E4M3(fn) bit pattern -> f32. exp>0: (1+m/8)*2^(e-7); exp==0: m*2^-9.
__device__ __forceinline__ float decode_e4m3(uint32_t b) {
  uint32_t sign = (b & 0x80u) << 24;
  uint32_t exp  = (b >> 3) & 0xFu;
  uint32_t man  = b & 0x7u;
  float mag;
  if (exp != 0u)
    mag = __uint_as_float(((exp + 120u) << 23) | (man << 20)); // e-7+127
  else
    mag = (float)man * 0.001953125f;                           // man * 2^-9
  union { float f; uint32_t i; } c; c.f = mag; c.i |= sign;
  return c.f;
}

union FragBF { v16bf v; uint4 q[2]; };
union Pack8  { uint16_t h[8]; uint4 q; };

// ---------------------------------------------------------------------------
// Pass 1: dequantize an N-chunk of W (fp8 bits as int32) to bf16 rows in d_ws.
// ---------------------------------------------------------------------------
__global__ __launch_bounds__(256)
void fp8lin_dequant(const int* __restrict__ Wq, const uint16_t* __restrict__ Wscale,
                    uint16_t* __restrict__ Wd, int n0, long long totalElems)
{
  long long idx = ((long long)blockIdx.x * 256 + threadIdx.x) * 8;
  if (idx >= totalElems) return;
  const int rowL = (int)(idx >> 12);                 // / K_DIM (4096)
  const float s = bf16_bits_to_f32(Wscale[n0 + rowL]);
  const int4* src = (const int4*)(Wq + (size_t)n0 * K_DIM + idx);
  int4 w0 = src[0], w1 = src[1];
  Pack8 t;
  t.h[0] = f32_to_bf16_bits(decode_e4m3((uint32_t)w0.x) * s);
  t.h[1] = f32_to_bf16_bits(decode_e4m3((uint32_t)w0.y) * s);
  t.h[2] = f32_to_bf16_bits(decode_e4m3((uint32_t)w0.z) * s);
  t.h[3] = f32_to_bf16_bits(decode_e4m3((uint32_t)w0.w) * s);
  t.h[4] = f32_to_bf16_bits(decode_e4m3((uint32_t)w1.x) * s);
  t.h[5] = f32_to_bf16_bits(decode_e4m3((uint32_t)w1.y) * s);
  t.h[6] = f32_to_bf16_bits(decode_e4m3((uint32_t)w1.z) * s);
  t.h[7] = f32_to_bf16_bits(decode_e4m3((uint32_t)w1.w) * s);
  *(uint4*)(Wd + idx) = t.q;
}

// ---------------------------------------------------------------------------
// Pass 2: bf16 WMMA GEMM with LDS double-buffering.
//   PREDEQ=true : B tiles come pre-dequantized from Wd (chunk-local rows);
//                 A and B tiles staged with GLOBAL_LOAD_ASYNC_TO_LDS_B128.
//   PREDEQ=false: B tiles decoded+scaled on the fly from Wq (fallback).
// Block tile 256x128x32; 16 waves, each wave 64x32 (4x2 WMMA accumulators).
// ---------------------------------------------------------------------------
template <bool PREDEQ>
__global__ __launch_bounds__(NTHREADS)
void fp8lin_wmma_gemm(const uint16_t* __restrict__ X,
                      const uint16_t* __restrict__ Wd,
                      const int*      __restrict__ Wq,
                      const uint16_t* __restrict__ Wscale,
                      const uint16_t* __restrict__ Bias,
                      uint16_t*       __restrict__ Out,
                      int n0)
{
  __shared__ __align__(16) uint16_t As[2][BM][LDS_STRIDE];
  __shared__ __align__(16) uint16_t Bs[2][BN][LDS_STRIDE];

  const int tid  = (int)threadIdx.x;
  const int lane = tid & 31;
  const int wave = tid >> 5;
  const int wm   = wave & 3;     // wave M index (x64)
  const int wn   = wave >> 2;    // wave N index (x32)

  const int m0  = (int)blockIdx.y * BM;
  const int nc0 = (int)blockIdx.x * BN;   // chunk-local N base
  const int ng0 = n0 + nc0;               // global N base

  // global -> LDS staging assignments
  const int aRow   = tid >> 1;           // 0..255
  const int aHalfK = (tid & 1) << 4;     // 0 / 16 halfwords
  const int bRow   = tid >> 2;           // 0..127
  const int bQK    = (tid & 3) << 3;     // 0/8/16/24 halfwords

  const uint16_t* aSrc  = X + (size_t)(m0 + aRow) * K_DIM + aHalfK;
  const uint16_t* bSrcD = nullptr;
  const int*      bSrcQ = nullptr;
  float bScale = 0.f;
  if (PREDEQ) {
    bSrcD = Wd + (size_t)(nc0 + bRow) * K_DIM + bQK;
  } else {
    bSrcQ = Wq + (size_t)(ng0 + bRow) * K_DIM + bQK;
    bScale = bf16_bits_to_f32(Wscale[ng0 + bRow]);
  }

  const v8f vzero = {0.f,0.f,0.f,0.f,0.f,0.f,0.f,0.f};
  v8f acc[4][2];
#pragma unroll
  for (int i = 0; i < 4; ++i)
#pragma unroll
    for (int j = 0; j < 2; ++j) acc[i][j] = vzero;

  // -------- prologue: stage K-tile 0 into buffer 0 --------
  {
#if HAVE_ASYNC_LDS
    // One address pair covers both 16B pieces: INST_OFFSET is added to both
    // the LDS and the global address.
    __builtin_amdgcn_global_load_async_to_lds_b128(
        (as1_v4i*)aSrc, (as3_v4i*)&As[0][aRow][aHalfK], 0, 0);
    __builtin_amdgcn_global_load_async_to_lds_b128(
        (as1_v4i*)aSrc, (as3_v4i*)&As[0][aRow][aHalfK], 16, 0);
    if (PREDEQ) {
      __builtin_amdgcn_global_load_async_to_lds_b128(
          (as1_v4i*)bSrcD, (as3_v4i*)&Bs[0][bRow][bQK], 0, 0);
    }
#else
    const uint4* s = (const uint4*)aSrc;
    uint4* dA = (uint4*)&As[0][aRow][aHalfK];
    dA[0] = s[0]; dA[1] = s[1];
    if (PREDEQ) {
      *(uint4*)&Bs[0][bRow][bQK] = *(const uint4*)bSrcD;
    }
#endif
    if (!PREDEQ) {
      const int4* sq = (const int4*)bSrcQ;
      int4 w0 = sq[0], w1 = sq[1];
      Pack8 t;
      t.h[0] = f32_to_bf16_bits(decode_e4m3((uint32_t)w0.x) * bScale);
      t.h[1] = f32_to_bf16_bits(decode_e4m3((uint32_t)w0.y) * bScale);
      t.h[2] = f32_to_bf16_bits(decode_e4m3((uint32_t)w0.z) * bScale);
      t.h[3] = f32_to_bf16_bits(decode_e4m3((uint32_t)w0.w) * bScale);
      t.h[4] = f32_to_bf16_bits(decode_e4m3((uint32_t)w1.x) * bScale);
      t.h[5] = f32_to_bf16_bits(decode_e4m3((uint32_t)w1.y) * bScale);
      t.h[6] = f32_to_bf16_bits(decode_e4m3((uint32_t)w1.z) * bScale);
      t.h[7] = f32_to_bf16_bits(decode_e4m3((uint32_t)w1.w) * bScale);
      *(uint4*)&Bs[0][bRow][bQK] = t.q;
    }
#if HAVE_ASYNC_LDS
    wait_async0();
#endif
  }
  __syncthreads();

#if !HAVE_ASYNC_LDS
  uint4 aReg0, aReg1;
  uint4 bRegD;
#endif
  int4 wReg0, wReg1;

  const int nK = K_DIM / BK;   // 128
  for (int kt = 0; kt < nK; ++kt) {
    const int  buf     = kt & 1;
    const int  nb      = buf ^ 1;
    const bool hasNext = (kt + 1) < nK;

    // Issue next-tile staging first (hidden behind the WMMA burst).
    if (hasNext) {
#if HAVE_ASYNC_LDS
      __builtin_amdgcn_global_load_async_to_lds_b128(
          (as1_v4i*)(aSrc + (kt + 1) * BK), (as3_v4i*)&As[nb][aRow][aHalfK], 0, 0);
      __builtin_amdgcn_global_load_async_to_lds_b128(
          (as1_v4i*)(aSrc + (kt + 1) * BK), (as3_v4i*)&As[nb][aRow][aHalfK], 16, 0);
      if (PREDEQ) {
        __builtin_amdgcn_global_load_async_to_lds_b128(
            (as1_v4i*)(bSrcD + (kt + 1) * BK), (as3_v4i*)&Bs[nb][bRow][bQK], 0, 0);
      } else {
        const int4* sq = (const int4*)(bSrcQ + (kt + 1) * BK);
        wReg0 = sq[0]; wReg1 = sq[1];
      }
#else
      const uint4* s = (const uint4*)(aSrc + (kt + 1) * BK);
      aReg0 = s[0]; aReg1 = s[1];
      if (PREDEQ) {
        bRegD = *(const uint4*)(bSrcD + (kt + 1) * BK);
      } else {
        const int4* sq = (const int4*)(bSrcQ + (kt + 1) * BK);
        wReg0 = sq[0]; wReg1 = sq[1];
      }
#endif
      if (kt + 2 < nK) {
        __builtin_prefetch(aSrc + (kt + 2) * BK, 0, 3);
        if (PREDEQ) __builtin_prefetch(bSrcD + (kt + 2) * BK, 0, 3);
        else        __builtin_prefetch(bSrcQ + (kt + 2) * BK, 0, 3);
      }
    }

    // Fragment loads per ISA VGPR layouts (16-bit A 16x32, B 32x16, wave32).
    FragBF a[4], b[2];
    const int fr = lane & 15;
    const int kA = (lane >> 4) << 3;   // A: lane half -> K 0/8 (+16 for v4..7)
    const int kB = (lane >> 4) << 3;   // B: lane half -> K 0/8 (+16 for v4..7)
#pragma unroll
    for (int i = 0; i < 4; ++i) {
      const uint16_t* p = &As[buf][wm * 64 + i * 16 + fr][0];
      a[i].q[0] = *(const uint4*)(p + kA);
      a[i].q[1] = *(const uint4*)(p + 16 + kA);
    }
#pragma unroll
    for (int j = 0; j < 2; ++j) {
      const uint16_t* p = &Bs[buf][wn * 32 + j * 16 + fr][0];
      b[j].q[0] = *(const uint4*)(p + kB);
      b[j].q[1] = *(const uint4*)(p + 16 + kB);
    }

#pragma unroll
    for (int i = 0; i < 4; ++i)
#pragma unroll
      for (int j = 0; j < 2; ++j)
        acc[i][j] = __builtin_amdgcn_wmma_f32_16x16x32_bf16(
            false, a[i].v, false, b[j].v, (short)0, acc[i][j], false, false);

    // Commit next tile to the other LDS buffer.
    if (hasNext) {
#if HAVE_ASYNC_LDS
      if (!PREDEQ) {
        Pack8 t;
        t.h[0] = f32_to_bf16_bits(decode_e4m3((uint32_t)wReg0.x) * bScale);
        t.h[1] = f32_to_bf16_bits(decode_e4m3((uint32_t)wReg0.y) * bScale);
        t.h[2] = f32_to_bf16_bits(decode_e4m3((uint32_t)wReg0.z) * bScale);
        t.h[3] = f32_to_bf16_bits(decode_e4m3((uint32_t)wReg0.w) * bScale);
        t.h[4] = f32_to_bf16_bits(decode_e4m3((uint32_t)wReg1.x) * bScale);
        t.h[5] = f32_to_bf16_bits(decode_e4m3((uint32_t)wReg1.y) * bScale);
        t.h[6] = f32_to_bf16_bits(decode_e4m3((uint32_t)wReg1.z) * bScale);
        t.h[7] = f32_to_bf16_bits(decode_e4m3((uint32_t)wReg1.w) * bScale);
        *(uint4*)&Bs[nb][bRow][bQK] = t.q;
      }
      wait_async0();     // this wave's async transfers into buffer nb are done
#else
      uint4* dA = (uint4*)&As[nb][aRow][aHalfK];
      dA[0] = aReg0; dA[1] = aReg1;
      uint4* dB = (uint4*)&Bs[nb][bRow][bQK];
      if (PREDEQ) {
        dB[0] = bRegD;
      } else {
        Pack8 t;
        t.h[0] = f32_to_bf16_bits(decode_e4m3((uint32_t)wReg0.x) * bScale);
        t.h[1] = f32_to_bf16_bits(decode_e4m3((uint32_t)wReg0.y) * bScale);
        t.h[2] = f32_to_bf16_bits(decode_e4m3((uint32_t)wReg0.z) * bScale);
        t.h[3] = f32_to_bf16_bits(decode_e4m3((uint32_t)wReg0.w) * bScale);
        t.h[4] = f32_to_bf16_bits(decode_e4m3((uint32_t)wReg1.x) * bScale);
        t.h[5] = f32_to_bf16_bits(decode_e4m3((uint32_t)wReg1.y) * bScale);
        t.h[6] = f32_to_bf16_bits(decode_e4m3((uint32_t)wReg1.z) * bScale);
        t.h[7] = f32_to_bf16_bits(decode_e4m3((uint32_t)wReg1.w) * bScale);
        dB[0] = t.q;
      }
#endif
    }
    __syncthreads();
  }

  // -------- epilogue: C/D layout = VGPR r -> row r (+8 for lane half 1) --------
  const int colLane = lane & 15;
  const int rowOff  = (lane >> 4) << 3;
#pragma unroll
  for (int j = 0; j < 2; ++j) {
    const int n = ng0 + wn * 32 + j * 16 + colLane;
    const float biasf = bf16_bits_to_f32(Bias[n]);
#pragma unroll
    for (int i = 0; i < 4; ++i) {
      const int mb = m0 + wm * 64 + i * 16 + rowOff;
#pragma unroll
      for (int r = 0; r < 8; ++r) {
        float v = acc[i][j][r];
        float o = bf16_bits_to_f32(f32_to_bf16_bits(v)) + biasf;  // bf16 GEMM out + bf16 bias
        Out[(size_t)(mb + r) * N_DIM + n] = f32_to_bf16_bits(o);
      }
    }
  }
}

// ---------------------------------------------------------------------------
extern "C" void kernel_launch(void* const* d_in, const int* in_sizes, int n_in,
                              void* d_out, int out_size, void* d_ws, size_t ws_size,
                              hipStream_t stream)
{
  (void)in_sizes; (void)n_in; (void)out_size;
  const uint16_t* X  = (const uint16_t*)d_in[0];  // x bf16 [M,K]
  const int*      Wq = (const int*)d_in[1];       // weight fp8 bits [N,K]
  const uint16_t* Ws = (const uint16_t*)d_in[2];  // weight_scale bf16 [N]
  const uint16_t* Bi = (const uint16_t*)d_in[3];  // bias bf16 [N]
  uint16_t* Out = (uint16_t*)d_out;               // out bf16 [M,N]
  uint16_t* Wd  = (uint16_t*)d_ws;

  // How many N-rows of dequantized bf16 weight fit in scratch (multiple of BN)?
  const size_t rowBytes = (size_t)K_DIM * sizeof(uint16_t);
  size_t maxRows = ws_size / rowBytes;
  int NS = (int)((maxRows / BN) * BN);
  if (NS > N_DIM) NS = N_DIM;

  if (NS >= BN) {
    // Two-pass, chunked over N by scratch capacity (stream order gives deps).
    for (int n0 = 0; n0 < N_DIM; n0 += NS) {
      int nc = N_DIM - n0; if (nc > NS) nc = NS;   // multiple of BN
      long long elems = (long long)nc * K_DIM;
      unsigned dqBlocks = (unsigned)((elems / 8 + 255) / 256);
      fp8lin_dequant<<<dim3(dqBlocks), dim3(256), 0, stream>>>(Wq, Ws, Wd, n0, elems);
      dim3 grid((unsigned)(nc / BN), M_DIM / BM);
      fp8lin_wmma_gemm<true><<<grid, dim3(NTHREADS), 0, stream>>>(
          X, Wd, nullptr, Ws, Bi, Out, n0);
    }
  } else {
    // Scratch too small: fused on-the-fly dequant fallback.
    dim3 grid(N_DIM / BN, M_DIM / BM);
    fp8lin_wmma_gemm<false><<<grid, dim3(NTHREADS), 0, stream>>>(
        X, nullptr, Wq, Ws, Bi, Out, 0);
  }
}
